// LSTMModel_33801392620228
// MI455X (gfx1250) — matrix operand order repaired and checked by reading, not compiled, not run on hardware
//
#include <hip/hip_runtime.h>
#include <hip/hip_bf16.h>
#include <stdint.h>

// ---------------------------------------------------------------------------
// Types for CDNA5 WMMA (wave32): v_wmma_f32_16x16x32_bf16
// ---------------------------------------------------------------------------
typedef __attribute__((ext_vector_type(16))) __bf16 v16bf;
typedef __attribute__((ext_vector_type(8)))  float  v8f;

union BF16x16 {
    uint4  q[2];
    v16bf  v;
};
union F32x8 {
    float4 f4[2];
    v8f    v;
};

__device__ __forceinline__ unsigned short f2bf(float f) {
    union { float f; uint32_t u; } cv; cv.f = f;
    uint32_t u = cv.u;
    u += 0x7fffu + ((u >> 16) & 1u);          // round-to-nearest-even
    return (unsigned short)(u >> 16);
}

__device__ __forceinline__ float fsig(float x) {
    return 1.0f / (1.0f + __expf(-x));
}

// ---------------------------------------------------------------------------
// Cast fp32 [M,K] -> bf16 [M,Kp] with zero pad (Kp % 32 == 0).
// If permToT != 0: output row r = t*64+b reads input row b*512+t
// (i.e. [B=64,T=512,*] -> [T,B,*] reorder, used for x only).
// ---------------------------------------------------------------------------
__global__ void cast_pad_f32_bf16(const float* __restrict__ in,
                                  unsigned short* __restrict__ out,
                                  int M, int K, int Kp, int permToT) {
    int i = blockIdx.x * blockDim.x + threadIdx.x;
    if (i >= M * Kp) return;
    int row = i / Kp, col = i - row * Kp;
    int srow = permToT ? ((row & 63) * 512 + (row >> 6)) : row;
    out[i] = (col < K) ? f2bf(in[(size_t)srow * K + col]) : (unsigned short)0;
}

// ---------------------------------------------------------------------------
// Tiled bf16 WMMA GEMM:  out = A[M,Kp] @ W[N,Kp]^T + bias1 + bias2
// A rows are t-major: m = t*64 + b.  Output stored in WMMA-fragment-swizzled
// layout: idx = ((((t*(N/16) + col16)*4 + rowtile)*2 + half)*16 + lane)*8 + e
// so producer stores and consumer loads are 32B-contiguous per lane.
// Requires M % 128 == 0, N % 64 == 0, Kp % 32 == 0.
// ---------------------------------------------------------------------------
#define ASTR 40   // padded LDS row stride in bf16 units (80B = 5*16B, aligned)

__global__ __launch_bounds__(256)
void gemm_bf16_wmma(const unsigned short* __restrict__ A,
                    const unsigned short* __restrict__ W,
                    const float* __restrict__ bias1,
                    const float* __restrict__ bias2,
                    float* __restrict__ out,
                    int M, int N, int Kp) {
    __shared__ unsigned short As[128 * ASTR];
    __shared__ unsigned short Ws[64 * ASTR];

    const int tid  = threadIdx.x;
    const int bm   = blockIdx.y * 128;
    const int bn   = blockIdx.x * 64;
    const int wave = tid >> 5;
    const int lane = tid & 31;
    const int n    = lane & 15;
    const int half = lane >> 4;

    // Staging coordinates (uint4 = 8 bf16; 4 uint4 per 32-wide row)
    const int ar0 = (tid * 2) >> 2,     aq0 = (tid * 2) & 3;     // A: 2 uint4/thread
    const int ar1 = (tid * 2 + 1) >> 2, aq1 = (tid * 2 + 1) & 3;
    const int wr  = tid >> 2,           wq  = tid & 3;           // W: 1 uint4/thread

    v8f acc[4] = {};

    const int nk = Kp >> 5;
    for (int kc = 0; kc < nk; ++kc) {
        const int k0 = kc * 32;
        uint4 a0 = *(const uint4*)(A + (size_t)(bm + ar0) * Kp + k0 + aq0 * 8);
        uint4 a1 = *(const uint4*)(A + (size_t)(bm + ar1) * Kp + k0 + aq1 * 8);
        uint4 w0 = *(const uint4*)(W + (size_t)(bn + wr)  * Kp + k0 + wq * 8);
        *(uint4*)&As[ar0 * ASTR + aq0 * 8] = a0;
        *(uint4*)&As[ar1 * ASTR + aq1 * 8] = a1;
        *(uint4*)&Ws[wr  * ASTR + wq  * 8] = w0;
        __syncthreads();

        // A fragment (16-bit A 16x32): lane row m=lane&15;
        // elems 0..7 = K 8*half..+7 ; elems 8..15 = K 16+8*half..+7
        BF16x16 af;
        const int arow = wave * 16 + n;
        af.q[0] = *(const uint4*)&As[arow * ASTR + half * 8];
        af.q[1] = *(const uint4*)&As[arow * ASTR + 16 + half * 8];

        #pragma unroll
        for (int ct = 0; ct < 4; ++ct) {
            // B fragment (16-bit B 32x16): lane col n; elems 0..7 = K 8*half..+7 ; elems 8..15 = K 16+8*half..+7
            BF16x16 bf_;
            const int brow = ct * 16 + n;
            bf_.q[0] = *(const uint4*)&Ws[brow * ASTR + half * 8];
            bf_.q[1] = *(const uint4*)&Ws[brow * ASTR + 16 + half * 8];
            acc[ct] = __builtin_amdgcn_wmma_f32_16x16x32_bf16(
                false, af.v, false, bf_.v, (short)0, acc[ct], false, false);
        }
        __syncthreads();
    }

    // Swizzled store: t = bm/64 + wave/4 ; rowtile = wave%4
    const int t     = (bm >> 6) + (wave >> 2);
    const int rtile = wave & 3;
    const int n16   = N >> 4;
    #pragma unroll
    for (int ct = 0; ct < 4; ++ct) {
        const int col  = bn + ct * 16 + n;
        const int ct2  = (bn >> 4) + ct;
        float bsum = (bias1 ? bias1[col] : 0.0f) + (bias2 ? bias2[col] : 0.0f);
        F32x8 s;
        #pragma unroll
        for (int e = 0; e < 8; ++e) s.v[e] = acc[ct][e] + bsum;
        float* p = out + ((((size_t)t * n16 + ct2) * 4 + rtile) * 2 + half) * 128 + n * 8;
        *(float4*)p       = s.f4[0];
        *(float4*)(p + 4) = s.f4[1];
    }
}

// ---------------------------------------------------------------------------
// Persistent LSTM scan. grid.x = H/64 workgroups; each WG owns 64 hidden
// units, W_hh slice resident in LDS (4 gates x 64 cols x H, bf16).
// Per step: gates = xg[t] (swizzled, vector-loaded) + h_prev @ Whh via WMMA,
// nonlinearities, double-buffered h, grid-wide barrier (atomic counter).
// ---------------------------------------------------------------------------
__global__ __launch_bounds__(256)
void lstm_scan(const float* __restrict__ xg,            // [T][4H/16][4][2][16][8]
               const unsigned short* __restrict__ Whh,  // [4H,H] bf16
               unsigned short* __restrict__ hbuf,       // [2,64,H] bf16
               unsigned short* __restrict__ seqout,     // [T,64,H] bf16 or null
               float* __restrict__ lastout,             // [64,H] f32 or null
               unsigned int* __restrict__ counter,
               int H, int T, int NWG) {
    extern __shared__ unsigned short Wlds[];            // [4*64, H]
    const int tid = threadIdx.x;
    const int j0  = blockIdx.x * 64;

    // Preload Whh slice: LDS row r = g*64+col holds Whh[g*H + j0 + col][0..H)
    {
        const int g   = tid >> 6;
        const int col = tid & 63;
        const unsigned short* src = Whh + (size_t)(g * H + j0 + col) * H;
        unsigned short* dst = Wlds + tid * H;
        for (int k = 0; k < H; k += 8)
            *(uint4*)(dst + k) = *(const uint4*)(src + k);
    }
    __syncthreads();

    const int wave = tid >> 5;
    const int lane = tid & 31;
    const int n    = lane & 15;
    const int half = lane >> 4;
    const int ct   = wave & 3;           // col-tile 0..3
    const int rg   = wave >> 2;          // row-group 0..1 (two 16-row tiles)
    const int colL = ct * 16 + n;
    const int colG = j0 + colL;
    const int h16  = H >> 4;
    const int stepElems = 256 * H;       // 64 * 4H floats per timestep

    v8f c[2] = {};                       // cell state, WMMA C/D lane layout

    for (int t = 0; t < T; ++t) {
        const unsigned short* hprev = hbuf + (size_t)(t & 1) * 64 * H;
        unsigned short*       hnext = hbuf + (size_t)((t + 1) & 1) * 64 * H;
        const float*          xgt   = xg + (size_t)t * stepElems;

        // Init accumulators from swizzled gate tensor: 2x b128 per fragment
        v8f acc[2][4];
        #pragma unroll
        for (int r = 0; r < 2; ++r) {
            const int rt = rg * 2 + r;
            #pragma unroll
            for (int g = 0; g < 4; ++g) {
                const int ct2 = g * h16 + (j0 >> 4) + ct;
                const float* p =
                    xgt + (((size_t)ct2 * 4 + rt) * 2 + half) * 128 + n * 8;
                F32x8 tmp;
                tmp.f4[0] = *(const float4*)p;
                tmp.f4[1] = *(const float4*)(p + 4);
                acc[r][g] = tmp.v;
            }
        }

        // Recurrent GEMM: h_prev[64,H] x Whh_slice -> 64x64 per gate
        for (int kc = 0; kc < H / 32; ++kc) {
            const int k0 = kc * 32;
            BF16x16 a0, a1;
            const int row0 = (rg * 2) * 16 + n;
            const int row1 = row0 + 16;
            a0.q[0] = *(const uint4*)(hprev + (size_t)row0 * H + k0 + half * 8);
            a0.q[1] = *(const uint4*)(hprev + (size_t)row0 * H + k0 + 16 + half * 8);
            a1.q[0] = *(const uint4*)(hprev + (size_t)row1 * H + k0 + half * 8);
            a1.q[1] = *(const uint4*)(hprev + (size_t)row1 * H + k0 + 16 + half * 8);
            #pragma unroll
            for (int g = 0; g < 4; ++g) {
                BF16x16 b;
                const unsigned short* wp =
                    Wlds + (g * 64 + colL) * H + k0 + half * 8;
                b.q[0] = *(const uint4*)(wp);
                b.q[1] = *(const uint4*)(wp + 16);
                acc[0][g] = __builtin_amdgcn_wmma_f32_16x16x32_bf16(
                    false, a0.v, false, b.v, (short)0, acc[0][g], false, false);
                acc[1][g] = __builtin_amdgcn_wmma_f32_16x16x32_bf16(
                    false, a1.v, false, b.v, (short)0, acc[1][g], false, false);
            }
        }

        // Gate nonlinearities + state update (PyTorch order i,f,g,o)
        #pragma unroll
        for (int r = 0; r < 2; ++r) {
            const int rt = rg * 2 + r;
            #pragma unroll
            for (int e = 0; e < 8; ++e) {
                float iv = fsig(acc[r][0][e]);
                float fv = fsig(acc[r][1][e]);
                float gv = tanhf(acc[r][2][e]);
                float ov = fsig(acc[r][3][e]);
                float cv = fv * c[r][e] + iv * gv;
                c[r][e] = cv;
                float hv = ov * tanhf(cv);
                const int row = rt * 16 + e + half * 8;
                unsigned short hb = f2bf(hv);
                hnext[(size_t)row * H + colG] = hb;
                if (seqout)
                    seqout[((size_t)t * 64 + row) * H + colG] = hb;
                if (lastout && t == T - 1)
                    lastout[(size_t)row * H + colG] = hv;
            }
        }

        // Grid-wide barrier (monotone counter, device scope)
        __threadfence();
        __syncthreads();
        if (tid == 0) {
            atomicAdd(counter, 1u);
            const unsigned target = (unsigned)(t + 1) * (unsigned)NWG;
            while (__hip_atomic_load(counter, __ATOMIC_RELAXED,
                                     __HIP_MEMORY_SCOPE_AGENT) < target) {
                __builtin_amdgcn_s_sleep(1);
            }
        }
        __syncthreads();
        __threadfence();
    }
}

// ---------------------------------------------------------------------------
// FC head (tiny; fp32 scalar)
// ---------------------------------------------------------------------------
__global__ void fc1_relu(const float* __restrict__ last,   // [64,256]
                         const float* __restrict__ W,      // [512,256]
                         const float* __restrict__ b,      // [512]
                         float* __restrict__ z) {          // [64,512]
    int idx = blockIdx.x * blockDim.x + threadIdx.x;
    if (idx >= 64 * 512) return;
    int row = idx >> 9, col = idx & 511;
    float s = b[col];
    const float* a = last + row * 256;
    const float* w = W + col * 256;
    for (int k = 0; k < 256; ++k) s += a[k] * w[k];
    z[idx] = s > 0.0f ? s : 0.0f;
}

__global__ void fc2k(const float* __restrict__ z,          // [64,512]
                     const float* __restrict__ W,          // [7,512]
                     const float* __restrict__ b,          // [7]
                     float* __restrict__ out) {            // [64,7]
    int idx = blockIdx.x * blockDim.x + threadIdx.x;
    if (idx >= 64 * 7) return;
    int row = idx / 7, col = idx % 7;
    float s = b[col];
    for (int k = 0; k < 512; ++k) s += z[row * 512 + k] * W[col * 512 + k];
    out[idx] = s;
}

// ---------------------------------------------------------------------------
// Launch
// ---------------------------------------------------------------------------
extern "C" void kernel_launch(void* const* d_in, const int* in_sizes, int n_in,
                              void* d_out, int out_size, void* d_ws, size_t ws_size,
                              hipStream_t stream) {
    (void)in_sizes; (void)n_in; (void)out_size; (void)ws_size;

    constexpr int Bn = 64, T = 512, D = 204;
    constexpr int Dp = 224;                 // D padded to multiple of 32
    constexpr int H1 = 512, H2 = 256;
    constexpr int G1 = 4 * H1, G2 = 4 * H2;
    constexpr int BT = Bn * T;

    const float* x     = (const float*)d_in[0];
    const float* Wih1  = (const float*)d_in[1];
    const float* Whh1  = (const float*)d_in[2];
    const float* bih1  = (const float*)d_in[3];
    const float* bhh1  = (const float*)d_in[4];
    const float* Wih2  = (const float*)d_in[5];
    const float* Whh2  = (const float*)d_in[6];
    const float* bih2  = (const float*)d_in[7];
    const float* bhh2  = (const float*)d_in[8];
    const float* Wfc1  = (const float*)d_in[9];
    const float* bfc1  = (const float*)d_in[10];
    const float* Wfc2  = (const float*)d_in[11];
    const float* bfc2  = (const float*)d_in[12];
    float* out = (float*)d_out;

    char* ws = (char*)d_ws;
    size_t off = 0;
    auto alloc = [&](size_t bytes) -> void* {
        void* p = ws + off;
        off += (bytes + 255) & ~(size_t)255;
        return p;
    };

    unsigned short* xb    = (unsigned short*)alloc((size_t)BT * Dp * 2);
    unsigned short* wih1b = (unsigned short*)alloc((size_t)G1 * Dp * 2);
    unsigned short* whh1b = (unsigned short*)alloc((size_t)G1 * H1 * 2);
    unsigned short* wih2b = (unsigned short*)alloc((size_t)G2 * H1 * 2);
    unsigned short* whh2b = (unsigned short*)alloc((size_t)G2 * H2 * 2);
    float*          xg1   = (float*)alloc((size_t)BT * G1 * 4);
    float*          xg2   = (float*)alloc((size_t)BT * G2 * 4);
    unsigned short* h1seq = (unsigned short*)alloc((size_t)BT * H1 * 2);
    // zero-init region (contiguous): hbuf1 | hbuf2 | counters
    unsigned short* hbuf1 = (unsigned short*)alloc((size_t)2 * Bn * H1 * 2);
    unsigned short* hbuf2 = (unsigned short*)alloc((size_t)2 * Bn * H2 * 2);
    unsigned int*   ctrs  = (unsigned int*)alloc(256);
    float*          h2l   = (float*)alloc((size_t)Bn * H2 * 4);
    float*          z     = (float*)alloc((size_t)Bn * 512 * 4);

    const size_t zeroBytes = (size_t)2 * Bn * H1 * 2 + (size_t)2 * Bn * H2 * 2 + 256;
    hipMemsetAsync(hbuf1, 0, zeroBytes, stream);

    // Casts to bf16 (with K padding; x additionally reordered [B,T,*]->[T,B,*])
    auto cast = [&](const float* src, unsigned short* dst, int M, int K, int Kp,
                    int perm) {
        int nElem = M * Kp;
        cast_pad_f32_bf16<<<(nElem + 255) / 256, 256, 0, stream>>>(
            src, dst, M, K, Kp, perm);
    };
    cast(x,    xb,    BT, D,  Dp, 1);
    cast(Wih1, wih1b, G1, D,  Dp, 0);
    cast(Whh1, whh1b, G1, H1, H1, 0);
    cast(Wih2, wih2b, G2, H1, H1, 0);
    cast(Whh2, whh2b, G2, H2, H2, 0);

    // Layer-1 input projection: [T*64,Dp] @ [2048,Dp]^T -> swizzled xg1
    gemm_bf16_wmma<<<dim3(G1 / 64, BT / 128), 256, 0, stream>>>(
        xb, wih1b, bih1, bhh1, xg1, BT, G1, Dp);

    // Layer-1 recurrent scan (8 WGs, 256KB LDS each)
    lstm_scan<<<dim3(H1 / 64), 256, (size_t)4 * 64 * H1 * 2, stream>>>(
        xg1, whh1b, hbuf1, h1seq, nullptr, ctrs, H1, T, H1 / 64);

    // Layer-2 input projection: [T*64,512] @ [1024,512]^T -> swizzled xg2
    gemm_bf16_wmma<<<dim3(G2 / 64, BT / 128), 256, 0, stream>>>(
        h1seq, wih2b, bih2, bhh2, xg2, BT, G2, H1);

    // Layer-2 recurrent scan (4 WGs, 128KB LDS each); only last h needed
    lstm_scan<<<dim3(H2 / 64), 256, (size_t)4 * 64 * H2 * 2, stream>>>(
        xg2, whh2b, hbuf2, nullptr, h2l, ctrs + 32, H2, T, H2 / 64);

    // FC head
    fc1_relu<<<(64 * 512 + 255) / 256, 256, 0, stream>>>(h2l, Wfc1, bfc1, z);
    fc2k<<<1, 512, 0, stream>>>(z, Wfc2, bfc2, out);
}
